// RouteNet_Fermi_79302276153372
// MI455X (gfx1250) — matrix operand order repaired and checked by reading, not compiled, not run on hardware
//
#include <hip/hip_runtime.h>
#include <hip/hip_bf16.h>

// ---------------------------------------------------------------------------
// RouteNet-Fermi for MI455X (gfx1250), wave32 + v_wmma_f32_16x16x32_bf16.
// One wave owns a 16-row tile. Weights are pre-swizzled into WMMA B-fragment
// order (frag-major, lane-major, element-contiguous) so every B fragment is
// two 16B vector loads; the compiler keeps them register-resident across the
// step loop and issues WMMAs back-to-back. Gather indices are software-
// pipelined one step ahead to hide the dependent index->row load chain.
// ---------------------------------------------------------------------------

constexpr int kNP  = 50000;
constexpr int kL   = 8;
constexpr int kNL  = 5000;
constexpr int kNQ  = 15000;
constexpr int kPPL = 80;
constexpr int kPPQ = 32;
constexpr int kQPL = 3;
constexpr int kIters = 8;

constexpr int PATH_TILES  = kNP / 16;          // 3125 (exact)
constexpr int QUEUE_TILES = (kNQ + 15) / 16;   // 938
constexpr int LINK_TILES  = (kNL + 15) / 16;   // 313

typedef __bf16 bfx16 __attribute__((ext_vector_type(16)));
typedef __bf16 bfx8  __attribute__((ext_vector_type(8)));
typedef float  fx8   __attribute__((ext_vector_type(8)));

__device__ __forceinline__ float sigmoidf_(float x) {
  return __builtin_amdgcn_rcpf(1.0f + __expf(-x));          // v_exp + v_rcp
}
__device__ __forceinline__ float tanhf_(float x) {
  // tanh(x) = 1 - 2/(exp(2x)+1); exp overflow/underflow saturate correctly.
  return 1.0f - 2.0f * __builtin_amdgcn_rcpf(__expf(2.0f * x) + 1.0f);
}

__device__ __forceinline__ fx8 wmma_bf16(bfx16 a, bfx16 b, fx8 c) {
  return __builtin_amdgcn_wmma_f32_16x16x32_bf16(false, a, false, b, (short)0, c, false, false);
}

__device__ __forceinline__ bfx16 join16(bfx8 lo, bfx8 hi) {
  return __builtin_shufflevector(lo, hi, 0, 1, 2, 3, 4, 5, 6, 7,
                                 8, 9, 10, 11, 12, 13, 14, 15);
}

// A-fragment (16x32 bf16, M x K) from a row-major 32-element row owned by this
// lane's M. ISA layout: lanes 0-15 hold K {0-7,16-23}; lanes 16-31 K {8-15,24-31}.
__device__ __forceinline__ bfx16 load_a_row32(const __bf16* row, int laneHi) {
  const bfx8* p0 = (const bfx8*)(row + 8 * laneHi);
  const bfx8* p1 = (const bfx8*)(row + 16 + 8 * laneHi);
  return join16(p0[0], p1[0]);
}

// B-fragment from pre-swizzled buffer: element (lane, e) at frag*512 + lane*16 + e.
__device__ __forceinline__ bfx16 load_b_frag(const __bf16* buf, int frag, int lane) {
  const bfx8* p = (const bfx8*)(buf + frag * 512 + lane * 16);
  return join16(p[0], p[1]);
}

// ---------------------------------------------------------------------------
// Weight conversion: f32 [K][128] row-major -> bf16 B-fragment-swizzled.
// Fragment (kb, nt) holds K rows kb*32..+31, cols nt*16..+15; within a fragment
// lane l = (laneHi,laneLo) owns (k = kb*32 + (e<8 ? 8*laneHi+e : 16+8*laneHi+e-8), n = nt*16 + laneLo).
// ---------------------------------------------------------------------------
__device__ __forceinline__ void swizzle_one(const float* __restrict__ W,
                                            __bf16* __restrict__ dst, int i) {
  int frag = i >> 9;          // 512 elements per fragment
  int rem  = i & 511;
  int lane = rem >> 4;
  int e    = rem & 15;
  int kb   = frag >> 3;
  int nt   = frag & 7;
  int laneLo = lane & 15, laneHi = lane >> 4;
  int k = kb * 32 + (e < 8 ? 8 * laneHi + e : 16 + 8 * laneHi + (e - 8));
  int n = nt * 16 + laneLo;
  dst[i] = (__bf16)W[k * 128 + n];
}

__global__ __launch_bounds__(256) void convert_weights_kernel(
    const float* __restrict__ pk, const float* __restrict__ pr,
    const float* __restrict__ qk, const float* __restrict__ qr,
    const float* __restrict__ lk, const float* __restrict__ lr,
    __bf16* __restrict__ dst) {
  int i = blockIdx.x * 256 + threadIdx.x;
  if (i < 8192)       swizzle_one(pk, dst, i);
  else if (i < 12288) swizzle_one(pr, dst + 8192, i - 8192);
  else if (i < 16384) swizzle_one(qk, dst + 12288, i - 12288);
  else if (i < 20480) swizzle_one(qr, dst + 16384, i - 16384);
  else if (i < 24576) swizzle_one(lk, dst + 20480, i - 20480);
  else if (i < 28672) swizzle_one(lr, dst + 24576, i - 24576);
}

// ---------------------------------------------------------------------------
// Embedding kernels (tiny MLPs, VALU is fine here)
// ---------------------------------------------------------------------------

__global__ __launch_bounds__(256) void path_embed_kernel(
    const float* __restrict__ traffic, const float* __restrict__ packets,
    const float* __restrict__ eq_lambda, const float* __restrict__ avg_pkts_lambda,
    const float* __restrict__ exp_max_factor, const float* __restrict__ pkts_lambda_on,
    const float* __restrict__ avg_t_off, const float* __restrict__ avg_t_on,
    const float* __restrict__ ar_a, const float* __restrict__ sigma,
    const int* __restrict__ model,
    const float* __restrict__ w1, const float* __restrict__ b1,
    const float* __restrict__ w2, const float* __restrict__ b2,
    float* __restrict__ path_h, float* __restrict__ path_c) {
  __shared__ float sw1[17 * 32], sb1[32], sw2[32 * 32], sb2[32];
  int tid = threadIdx.x;
  for (int i = tid; i < 17 * 32; i += 256) sw1[i] = w1[i];
  for (int i = tid; i < 32 * 32; i += 256) sw2[i] = w2[i];
  if (tid < 32) { sb1[tid] = b1[tid]; sb2[tid] = b2[tid]; }
  __syncthreads();
  int p = blockIdx.x * 256 + tid;
  if (p >= kNP) return;
  float f[17];
  f[0] = (traffic[p] - 1385.4058837890625f) / 859.8118896484375f;
  f[1] = (packets[p] - 1.4015231132507324f) / 0.8932565450668335f;
  int m = model[p];
#pragma unroll
  for (int k = 0; k < 7; ++k) f[2 + k] = (m == k) ? 1.0f : 0.0f;
  f[9]  = (eq_lambda[p] - 1350.97119140625f) / 858.316162109375f;
  f[10] = (avg_pkts_lambda[p] - 0.9117304086685181f) / 0.9723503589630127f;
  f[11] = (exp_max_factor[p] - 6.663637638092041f) / 4.715115070343018f;
  f[12] = (pkts_lambda_on[p] - 0.9116322994232178f) / 1.651275396347046f;
  f[13] = (avg_t_off[p] - 1.6649284362792969f) / 2.356407403945923f;
  f[14] = (avg_t_on[p] - 1.6649284362792969f) / 2.356407403945923f;
  f[15] = ar_a[p];
  f[16] = sigma[p];
  float h1[32];
#pragma unroll 4
  for (int d = 0; d < 32; ++d) {
    float acc = sb1[d];
    for (int k = 0; k < 17; ++k) acc += f[k] * sw1[k * 32 + d];
    h1[d] = fmaxf(acc, 0.0f);
  }
#pragma unroll 4
  for (int d = 0; d < 32; ++d) {
    float acc = sb2[d];
    for (int k = 0; k < 32; ++k) acc += h1[k] * sw2[k * 32 + d];
    path_h[p * 32 + d] = fmaxf(acc, 0.0f);
    path_c[p * 32 + d] = 0.0f;
  }
}

__global__ __launch_bounds__(256) void link_embed_kernel(
    const float* __restrict__ traffic, const float* __restrict__ capacity,
    const int* __restrict__ policy, const int* __restrict__ p2l,  // [NL, PPL, 2]
    const float* __restrict__ w1, const float* __restrict__ b1,
    const float* __restrict__ w2, const float* __restrict__ b2,
    float* __restrict__ link_h, float* __restrict__ link_c,
    __bf16* __restrict__ link_hb) {
  __shared__ float sw1[5 * 32], sb1[32], sw2[32 * 32], sb2[32];
  int tid = threadIdx.x;
  for (int i = tid; i < 5 * 32; i += 256) sw1[i] = w1[i];
  for (int i = tid; i < 32 * 32; i += 256) sw2[i] = w2[i];
  if (tid < 32) { sb1[tid] = b1[tid]; sb2[tid] = b2[tid]; }
  __syncthreads();
  int l = blockIdx.x * 256 + tid;
  if (l >= kNL) return;
  float load = 0.0f;
  const int* row = p2l + (size_t)l * kPPL * 2;
  for (int j = 0; j < kPPL; ++j) load += traffic[row[j * 2 + 0]];
  load /= capacity[l];
  int pol = policy[l];
  float f[5];
  f[0] = load;
#pragma unroll
  for (int k = 0; k < 4; ++k) f[1 + k] = (pol == k) ? 1.0f : 0.0f;
  float h1[32];
#pragma unroll 4
  for (int d = 0; d < 32; ++d) {
    float acc = sb1[d];
    for (int k = 0; k < 5; ++k) acc += f[k] * sw1[k * 32 + d];
    h1[d] = fmaxf(acc, 0.0f);
  }
#pragma unroll 4
  for (int d = 0; d < 32; ++d) {
    float acc = sb2[d];
    for (int k = 0; k < 32; ++k) acc += h1[k] * sw2[k * 32 + d];
    float h = fmaxf(acc, 0.0f);
    link_h[l * 32 + d] = h;
    link_c[l * 32 + d] = 0.0f;
    link_hb[l * 32 + d] = (__bf16)h;
  }
}

__global__ __launch_bounds__(256) void queue_embed_kernel(
    const float* __restrict__ queue_size, const float* __restrict__ weight,
    const int* __restrict__ priority,
    const float* __restrict__ w1, const float* __restrict__ b1,
    const float* __restrict__ w2, const float* __restrict__ b2,
    float* __restrict__ queue_h, float* __restrict__ queue_c,
    __bf16* __restrict__ queue_hb) {
  __shared__ float sw1[5 * 32], sb1[32], sw2[32 * 32], sb2[32];
  int tid = threadIdx.x;
  for (int i = tid; i < 5 * 32; i += 256) sw1[i] = w1[i];
  for (int i = tid; i < 32 * 32; i += 256) sw2[i] = w2[i];
  if (tid < 32) { sb1[tid] = b1[tid]; sb2[tid] = b2[tid]; }
  __syncthreads();
  int q = blockIdx.x * 256 + tid;
  if (q >= kNQ) return;
  int pr = priority[q];
  float f[5];
  f[0] = (queue_size[q] - 30259.10546875f) / 21410.095703125f;
#pragma unroll
  for (int k = 0; k < 3; ++k) f[1 + k] = (pr == k) ? 1.0f : 0.0f;
  f[4] = weight[q];
  float h1[32];
#pragma unroll 4
  for (int d = 0; d < 32; ++d) {
    float acc = sb1[d];
    for (int k = 0; k < 5; ++k) acc += f[k] * sw1[k * 32 + d];
    h1[d] = fmaxf(acc, 0.0f);
  }
#pragma unroll 4
  for (int d = 0; d < 32; ++d) {
    float acc = sb2[d];
    for (int k = 0; k < 32; ++k) acc += h1[k] * sw2[k * 32 + d];
    float h = fmaxf(acc, 0.0f);
    queue_h[q * 32 + d] = h;
    queue_c[q * 32 + d] = 0.0f;
    queue_hb[q * 32 + d] = (__bf16)h;
  }
}

// ---------------------------------------------------------------------------
// Path LSTM: 8 steps, z = [qg|lg] @ Wk(64x128) + h @ Wr(32x128) + b.
// 24 WMMAs per step per wave (8 N-tiles x 3 K-stages).
// Pre-swizzled fragments: Wk kb=0 -> frag nt, kb=1 -> frag 8+nt; Wr -> frag nt.
// Gather indices for step s+1 are loaded during step s.
// ---------------------------------------------------------------------------
__global__ __launch_bounds__(256) void path_lstm_kernel(
    const int* __restrict__ q2p, const int* __restrict__ l2p,
    const __bf16* __restrict__ wpk, const __bf16* __restrict__ wpr,
    const float* __restrict__ pb,
    const __bf16* __restrict__ queue_hb, const __bf16* __restrict__ link_hb,
    float* __restrict__ path_h, float* __restrict__ path_c,
    __bf16* __restrict__ seq) {
  __shared__ __attribute__((aligned(16))) __bf16 s_wk[64 * 128];
  __shared__ __attribute__((aligned(16))) __bf16 s_wr[32 * 128];
  __shared__ float s_b[128];
  __shared__ __attribute__((aligned(16))) __bf16 s_h[8][16 * 32];

  const int tid = threadIdx.x;
  for (int i = tid; i < 64 * 128; i += 256) s_wk[i] = wpk[i];
  for (int i = tid; i < 32 * 128; i += 256) s_wr[i] = wpr[i];
  if (tid < 128) s_b[tid] = pb[tid];
  __syncthreads();

  const int lane = tid & 31, wave = tid >> 5;
  const int laneLo = lane & 15, laneHi = lane >> 4;
  int tile = blockIdx.x * 8 + wave;
  const bool active = tile < PATH_TILES;
  if (!active) tile = PATH_TILES - 1;

  float bias[8];
#pragma unroll
  for (int nt = 0; nt < 8; ++nt) bias[nt] = s_b[nt * 16 + laneLo];

  fx8 hC[2], cC[2];
  const int rowBase = tile * 16 + 8 * laneHi;
#pragma unroll
  for (int t = 0; t < 2; ++t)
#pragma unroll
    for (int r = 0; r < 8; ++r) {
      int p = rowBase + r;
      hC[t][r] = path_h[p * 32 + t * 16 + laneLo];
      cC[t][r] = path_c[p * 32 + t * 16 + laneLo];
    }

  __bf16* sh = &s_h[wave][0];
#pragma unroll
  for (int t = 0; t < 2; ++t)
#pragma unroll
    for (int r = 0; r < 8; ++r)
      sh[(8 * laneHi + r) * 32 + t * 16 + laneLo] = (__bf16)hC[t][r];

  // Cooperative row copy LDS -> seq: lane handles row lane>>1, half lane&1.
  const int cm = lane >> 1, chf = lane & 1;
  const bfx8* csrc = (const bfx8*)(sh + cm * 32 + chf * 16);
  bfx8* cdst0 = (bfx8*)(seq + ((size_t)(tile * 16 + cm) * 9) * 32 + chf * 16);
  if (active) { cdst0[0] = csrc[0]; cdst0[1] = csrc[1]; }

  const int pA = tile * 16 + laneLo;  // A-fragment row owner
  int qi = q2p[pA * kL + 0];
  int li = l2p[pA * kL + 0];
#pragma unroll 1
  for (int s = 0; s < kL; ++s) {
    // Software-pipelined gather indices for step s+1.
    int qiN = qi, liN = li;
    if (s + 1 < kL) {
      qiN = q2p[pA * kL + s + 1];
      liN = l2p[pA * kL + s + 1];
    }
    bfx16 aQ = load_a_row32(queue_hb + (size_t)qi * 32, laneHi);
    bfx16 aL = load_a_row32(link_hb + (size_t)li * 32, laneHi);
    bfx16 aH = load_a_row32(sh + laneLo * 32, laneHi);  // DS in-order per wave

    fx8 z[8];
#pragma unroll
    for (int nt = 0; nt < 8; ++nt) {
      fx8 acc;
#pragma unroll
      for (int r = 0; r < 8; ++r) acc[r] = bias[nt];
      acc = wmma_bf16(aQ, load_b_frag(s_wk, nt, lane), acc);
      acc = wmma_bf16(aL, load_b_frag(s_wk, 8 + nt, lane), acc);
      acc = wmma_bf16(aH, load_b_frag(s_wr, nt, lane), acc);
      z[nt] = acc;
    }
#pragma unroll
    for (int t = 0; t < 2; ++t)
#pragma unroll
      for (int r = 0; r < 8; ++r) {
        float iv = sigmoidf_(z[0 + t][r]);
        float fv = sigmoidf_(z[2 + t][r]);
        float gv = tanhf_(z[4 + t][r]);
        float ov = sigmoidf_(z[6 + t][r]);
        float cn = fv * cC[t][r] + iv * gv;
        float hn = ov * tanhf_(cn);
        cC[t][r] = cn;
        hC[t][r] = hn;
        sh[(8 * laneHi + r) * 32 + t * 16 + laneLo] = (__bf16)hn;
      }
    // Cooperative store of this step's h tile into seq slot s+1.
    bfx8* cdst = (bfx8*)(seq + ((size_t)(tile * 16 + cm) * 9 + (s + 1)) * 32 + chf * 16);
    if (active) { cdst[0] = csrc[0]; cdst[1] = csrc[1]; }
    qi = qiN;
    li = liN;
  }
  if (active) {
#pragma unroll
    for (int t = 0; t < 2; ++t)
#pragma unroll
      for (int r = 0; r < 8; ++r) {
        int p = rowBase + r;
        path_h[p * 32 + t * 16 + laneLo] = hC[t][r];
        path_c[p * 32 + t * 16 + laneLo] = cC[t][r];
      }
  }
}

// ---------------------------------------------------------------------------
// Queue update: x = sum over 32 gathered seq rows, one LSTM cell.
// ---------------------------------------------------------------------------
__global__ __launch_bounds__(256) void queue_update_kernel(
    const int* __restrict__ p2q,  // [NQ, PPQ, 2]
    const __bf16* __restrict__ seq,
    const __bf16* __restrict__ wqk, const __bf16* __restrict__ wqr,
    const float* __restrict__ qb,
    float* __restrict__ queue_h, float* __restrict__ queue_c,
    __bf16* __restrict__ queue_hb) {
  __shared__ __attribute__((aligned(16))) __bf16 s_wk[32 * 128];
  __shared__ __attribute__((aligned(16))) __bf16 s_wr[32 * 128];
  __shared__ float s_b[128];
  __shared__ __attribute__((aligned(16))) __bf16 s_x[8][16 * 32];

  const int tid = threadIdx.x;
  for (int i = tid; i < 32 * 128; i += 256) { s_wk[i] = wqk[i]; s_wr[i] = wqr[i]; }
  if (tid < 128) s_b[tid] = qb[tid];
  __syncthreads();

  const int lane = tid & 31, wave = tid >> 5;
  const int laneLo = lane & 15, laneHi = lane >> 4;
  int tile = blockIdx.x * 8 + wave;
  const bool active = tile < QUEUE_TILES;
  if (!active) tile = QUEUE_TILES - 1;

  float bias[8];
#pragma unroll
  for (int nt = 0; nt < 8; ++nt) bias[nt] = s_b[nt * 16 + laneLo];

  int qA = tile * 16 + laneLo;
  if (qA >= kNQ) qA = kNQ - 1;

  // Lane owns row laneLo, dim half laneHi*16 .. +15 for the gather-sum.
  float xs[16];
#pragma unroll
  for (int d = 0; d < 16; ++d) xs[d] = 0.0f;
  const int* prow = p2q + (size_t)qA * kPPQ * 2;
  int pj = prow[0], slj = prow[1];
#pragma unroll 1
  for (int j = 0; j < kPPQ; ++j) {
    int pN = pj, slN = slj;
    if (j + 1 < kPPQ) { pN = prow[(j + 1) * 2]; slN = prow[(j + 1) * 2 + 1]; }
    const bfx8* src = (const bfx8*)(seq + ((size_t)pj * 9 + slj) * 32 + laneHi * 16);
    bfx8 v0 = src[0], v1 = src[1];
#pragma unroll
    for (int d = 0; d < 8; ++d) { xs[d] += (float)v0[d]; xs[8 + d] += (float)v1[d]; }
    pj = pN;
    slj = slN;
  }
  __bf16* sx = &s_x[wave][0];
#pragma unroll
  for (int d = 0; d < 16; ++d) sx[laneLo * 32 + laneHi * 16 + d] = (__bf16)xs[d];

  bfx16 aX = load_a_row32(sx + laneLo * 32, laneHi);
  bfx16 aH = load_a_row32(queue_hb + (size_t)qA * 32, laneHi);

  const int rowBase = tile * 16 + 8 * laneHi;
  fx8 hC[2], cC[2];
#pragma unroll
  for (int t = 0; t < 2; ++t)
#pragma unroll
    for (int r = 0; r < 8; ++r) {
      int q = rowBase + r;
      int qc = q < kNQ ? q : kNQ - 1;
      hC[t][r] = queue_h[qc * 32 + t * 16 + laneLo];
      cC[t][r] = queue_c[qc * 32 + t * 16 + laneLo];
    }

  fx8 z[8];
#pragma unroll
  for (int nt = 0; nt < 8; ++nt) {
    fx8 acc;
#pragma unroll
    for (int r = 0; r < 8; ++r) acc[r] = bias[nt];
    acc = wmma_bf16(aX, load_b_frag(s_wk, nt, lane), acc);
    acc = wmma_bf16(aH, load_b_frag(s_wr, nt, lane), acc);
    z[nt] = acc;
  }
#pragma unroll
  for (int t = 0; t < 2; ++t)
#pragma unroll
    for (int r = 0; r < 8; ++r) {
      int q = rowBase + r;
      float iv = sigmoidf_(z[0 + t][r]);
      float fv = sigmoidf_(z[2 + t][r]);
      float gv = tanhf_(z[4 + t][r]);
      float ov = sigmoidf_(z[6 + t][r]);
      float cn = fv * cC[t][r] + iv * gv;
      float hn = ov * tanhf_(cn);
      if (active && q < kNQ) {
        queue_h[q * 32 + t * 16 + laneLo] = hn;
        queue_c[q * 32 + t * 16 + laneLo] = cn;
        queue_hb[q * 32 + t * 16 + laneLo] = (__bf16)hn;
      }
    }
}

// ---------------------------------------------------------------------------
// Link LSTM: 3 steps over gathered queue states.
// ---------------------------------------------------------------------------
__global__ __launch_bounds__(256) void link_lstm_kernel(
    const int* __restrict__ q2l,  // [NL, QPL]
    const __bf16* __restrict__ queue_hb,
    const __bf16* __restrict__ wlk, const __bf16* __restrict__ wlr,
    const float* __restrict__ lb,
    float* __restrict__ link_h, float* __restrict__ link_c,
    __bf16* __restrict__ link_hb) {
  __shared__ __attribute__((aligned(16))) __bf16 s_wk[32 * 128];
  __shared__ __attribute__((aligned(16))) __bf16 s_wr[32 * 128];
  __shared__ float s_b[128];
  __shared__ __attribute__((aligned(16))) __bf16 s_h[8][16 * 32];

  const int tid = threadIdx.x;
  for (int i = tid; i < 32 * 128; i += 256) { s_wk[i] = wlk[i]; s_wr[i] = wlr[i]; }
  if (tid < 128) s_b[tid] = lb[tid];
  __syncthreads();

  const int lane = tid & 31, wave = tid >> 5;
  const int laneLo = lane & 15, laneHi = lane >> 4;
  int tile = blockIdx.x * 8 + wave;
  const bool active = tile < LINK_TILES;
  if (!active) tile = LINK_TILES - 1;

  float bias[8];
#pragma unroll
  for (int nt = 0; nt < 8; ++nt) bias[nt] = s_b[nt * 16 + laneLo];

  int lA = tile * 16 + laneLo;
  if (lA >= kNL) lA = kNL - 1;

  const int rowBase = tile * 16 + 8 * laneHi;
  fx8 hC[2], cC[2];
#pragma unroll
  for (int t = 0; t < 2; ++t)
#pragma unroll
    for (int r = 0; r < 8; ++r) {
      int l = rowBase + r;
      int lc = l < kNL ? l : kNL - 1;
      hC[t][r] = link_h[lc * 32 + t * 16 + laneLo];
      cC[t][r] = link_c[lc * 32 + t * 16 + laneLo];
    }

  __bf16* sh = &s_h[wave][0];
#pragma unroll
  for (int t = 0; t < 2; ++t)
#pragma unroll
    for (int r = 0; r < 8; ++r)
      sh[(8 * laneHi + r) * 32 + t * 16 + laneLo] = (__bf16)hC[t][r];

  int qi = q2l[lA * kQPL + 0];
#pragma unroll 1
  for (int s = 0; s < kQPL; ++s) {
    int qiN = qi;
    if (s + 1 < kQPL) qiN = q2l[lA * kQPL + s + 1];
    bfx16 aX = load_a_row32(queue_hb + (size_t)qi * 32, laneHi);
    bfx16 aH = load_a_row32(sh + laneLo * 32, laneHi);
    fx8 z[8];
#pragma unroll
    for (int nt = 0; nt < 8; ++nt) {
      fx8 acc;
#pragma unroll
      for (int r = 0; r < 8; ++r) acc[r] = bias[nt];
      acc = wmma_bf16(aX, load_b_frag(s_wk, nt, lane), acc);
      acc = wmma_bf16(aH, load_b_frag(s_wr, nt, lane), acc);
      z[nt] = acc;
    }
#pragma unroll
    for (int t = 0; t < 2; ++t)
#pragma unroll
      for (int r = 0; r < 8; ++r) {
        float iv = sigmoidf_(z[0 + t][r]);
        float fv = sigmoidf_(z[2 + t][r]);
        float gv = tanhf_(z[4 + t][r]);
        float ov = sigmoidf_(z[6 + t][r]);
        float cn = fv * cC[t][r] + iv * gv;
        float hn = ov * tanhf_(cn);
        cC[t][r] = cn;
        hC[t][r] = hn;
        sh[(8 * laneHi + r) * 32 + t * 16 + laneLo] = (__bf16)hn;
      }
    qi = qiN;
  }
#pragma unroll
  for (int t = 0; t < 2; ++t)
#pragma unroll
    for (int r = 0; r < 8; ++r) {
      int l = rowBase + r;
      if (active && l < kNL) {
        link_h[l * 32 + t * 16 + laneLo] = hC[t][r];
        link_c[l * 32 + t * 16 + laneLo] = cC[t][r];
        link_hb[l * 32 + t * 16 + laneLo] = (__bf16)hC[t][r];
      }
    }
}

// ---------------------------------------------------------------------------
// Readout MLP: 32 -> 16 -> 16 -> 1, sigmoid.
// ---------------------------------------------------------------------------
__global__ __launch_bounds__(256) void readout_kernel(
    const float* __restrict__ path_h,
    const float* __restrict__ w1, const float* __restrict__ b1,
    const float* __restrict__ w2, const float* __restrict__ b2,
    const float* __restrict__ w3, const float* __restrict__ b3,
    float* __restrict__ out) {
  __shared__ float sw1[32 * 16], sb1[16], sw2[16 * 16], sb2[16], sw3[16], sb3;
  int tid = threadIdx.x;
  for (int i = tid; i < 32 * 16; i += 256) sw1[i] = w1[i];
  for (int i = tid; i < 16 * 16; i += 256) sw2[i] = w2[i];
  if (tid < 16) { sb1[tid] = b1[tid]; sb2[tid] = b2[tid]; sw3[tid] = w3[tid]; }
  if (tid == 0) sb3 = b3[0];
  __syncthreads();
  int p = blockIdx.x * 256 + tid;
  if (p >= kNP) return;
  float x[32];
#pragma unroll
  for (int d = 0; d < 32; ++d) x[d] = path_h[p * 32 + d];
  float h1[16];
#pragma unroll 4
  for (int d = 0; d < 16; ++d) {
    float acc = sb1[d];
    for (int k = 0; k < 32; ++k) acc += x[k] * sw1[k * 16 + d];
    h1[d] = fmaxf(acc, 0.0f);
  }
  float h2[16];
#pragma unroll 4
  for (int d = 0; d < 16; ++d) {
    float acc = sb2[d];
    for (int k = 0; k < 16; ++k) acc += h1[k] * sw2[k * 16 + d];
    h2[d] = fmaxf(acc, 0.0f);
  }
  float acc = sb3;
#pragma unroll
  for (int k = 0; k < 16; ++k) acc += h2[k] * sw3[k];
  out[p] = sigmoidf_(acc);
}

// ---------------------------------------------------------------------------
// Workspace layout
// ---------------------------------------------------------------------------
constexpr size_t OFF_W   = 0;                                   // 28672 bf16
constexpr size_t OFF_PH  = 57344;                               // NP*32 f32
constexpr size_t OFF_PC  = OFF_PH + (size_t)kNP * 32 * 4;
constexpr size_t OFF_QH  = OFF_PC + (size_t)kNP * 32 * 4;
constexpr size_t OFF_QC  = OFF_QH + (size_t)kNQ * 32 * 4;
constexpr size_t OFF_QHB = OFF_QC + (size_t)kNQ * 32 * 4;
constexpr size_t OFF_LH  = OFF_QHB + (size_t)kNQ * 32 * 2;
constexpr size_t OFF_LC  = OFF_LH + (size_t)kNL * 32 * 4;
constexpr size_t OFF_LHB = OFF_LC + (size_t)kNL * 32 * 4;
constexpr size_t OFF_SEQ = OFF_LHB + (size_t)kNL * 32 * 2;      // NP*9*32 bf16

extern "C" void kernel_launch(void* const* d_in, const int* in_sizes, int n_in,
                              void* d_out, int out_size, void* d_ws, size_t ws_size,
                              hipStream_t stream) {
  (void)in_sizes; (void)n_in; (void)out_size; (void)ws_size;

  const float* traffic         = (const float*)d_in[0];
  const float* packets         = (const float*)d_in[1];
  const float* eq_lambda       = (const float*)d_in[2];
  const float* avg_pkts_lambda = (const float*)d_in[3];
  const float* exp_max_factor  = (const float*)d_in[4];
  const float* pkts_lambda_on  = (const float*)d_in[5];
  const float* avg_t_off       = (const float*)d_in[6];
  const float* avg_t_on        = (const float*)d_in[7];
  const float* ar_a            = (const float*)d_in[8];
  const float* sigma           = (const float*)d_in[9];
  const float* capacity        = (const float*)d_in[10];
  const float* queue_size      = (const float*)d_in[11];
  const float* weight          = (const float*)d_in[12];
  const int*   model           = (const int*)d_in[13];
  const int*   policy          = (const int*)d_in[14];
  const int*   priority        = (const int*)d_in[15];
  const int*   q2p             = (const int*)d_in[16];
  const int*   l2p             = (const int*)d_in[17];
  const int*   p2l             = (const int*)d_in[18];
  const int*   p2q             = (const int*)d_in[19];
  const int*   q2l             = (const int*)d_in[20];
  // params dict flattened in _make_params insertion order:
  const float* pe_w1 = (const float*)d_in[21];
  const float* pe_b1 = (const float*)d_in[22];
  const float* pe_w2 = (const float*)d_in[23];
  const float* pe_b2 = (const float*)d_in[24];
  const float* le_w1 = (const float*)d_in[25];
  const float* le_b1 = (const float*)d_in[26];
  const float* le_w2 = (const float*)d_in[27];
  const float* le_b2 = (const float*)d_in[28];
  const float* qe_w1 = (const float*)d_in[29];
  const float* qe_b1 = (const float*)d_in[30];
  const float* qe_w2 = (const float*)d_in[31];
  const float* qe_b2 = (const float*)d_in[32];
  const float* p_k   = (const float*)d_in[33];
  const float* p_r   = (const float*)d_in[34];
  const float* p_b   = (const float*)d_in[35];
  const float* q_k   = (const float*)d_in[36];
  const float* q_r   = (const float*)d_in[37];
  const float* q_b   = (const float*)d_in[38];
  const float* l_k   = (const float*)d_in[39];
  const float* l_r   = (const float*)d_in[40];
  const float* l_b   = (const float*)d_in[41];
  const float* ro_w1 = (const float*)d_in[42];
  const float* ro_b1 = (const float*)d_in[43];
  const float* ro_w2 = (const float*)d_in[44];
  const float* ro_b2 = (const float*)d_in[45];
  const float* ro_w3 = (const float*)d_in[46];
  const float* ro_b3 = (const float*)d_in[47];

  char* ws = (char*)d_ws;
  __bf16* wb       = (__bf16*)(ws + OFF_W);
  __bf16* wpk      = wb;
  __bf16* wpr      = wb + 8192;
  __bf16* wqk      = wb + 12288;
  __bf16* wqr      = wb + 16384;
  __bf16* wlk      = wb + 20480;
  __bf16* wlr      = wb + 24576;
  float*  path_h   = (float*)(ws + OFF_PH);
  float*  path_c   = (float*)(ws + OFF_PC);
  float*  queue_h  = (float*)(ws + OFF_QH);
  float*  queue_c  = (float*)(ws + OFF_QC);
  __bf16* queue_hb = (__bf16*)(ws + OFF_QHB);
  float*  link_h   = (float*)(ws + OFF_LH);
  float*  link_c   = (float*)(ws + OFF_LC);
  __bf16* link_hb  = (__bf16*)(ws + OFF_LHB);
  __bf16* seq      = (__bf16*)(ws + OFF_SEQ);
  float*  out      = (float*)d_out;

  convert_weights_kernel<<<(28672 + 255) / 256, 256, 0, stream>>>(
      p_k, p_r, q_k, q_r, l_k, l_r, wb);
  path_embed_kernel<<<(kNP + 255) / 256, 256, 0, stream>>>(
      traffic, packets, eq_lambda, avg_pkts_lambda, exp_max_factor,
      pkts_lambda_on, avg_t_off, avg_t_on, ar_a, sigma, model,
      pe_w1, pe_b1, pe_w2, pe_b2, path_h, path_c);
  link_embed_kernel<<<(kNL + 255) / 256, 256, 0, stream>>>(
      traffic, capacity, policy, p2l, le_w1, le_b1, le_w2, le_b2,
      link_h, link_c, link_hb);
  queue_embed_kernel<<<(kNQ + 255) / 256, 256, 0, stream>>>(
      queue_size, weight, priority, qe_w1, qe_b1, qe_w2, qe_b2,
      queue_h, queue_c, queue_hb);

  const int gridP = (PATH_TILES + 7) / 8;    // 391
  const int gridQ = (QUEUE_TILES + 7) / 8;   // 118
  const int gridL = (LINK_TILES + 7) / 8;    // 40
  for (int it = 0; it < kIters; ++it) {
    path_lstm_kernel<<<gridP, 256, 0, stream>>>(
        q2p, l2p, wpk, wpr, p_b, queue_hb, link_hb, path_h, path_c, seq);
    queue_update_kernel<<<gridQ, 256, 0, stream>>>(
        p2q, seq, wqk, wqr, q_b, queue_h, queue_c, queue_hb);
    link_lstm_kernel<<<gridL, 256, 0, stream>>>(
        q2l, queue_hb, wlk, wlr, l_b, link_h, link_c, link_hb);
  }
  readout_kernel<<<(kNP + 255) / 256, 256, 0, stream>>>(
      path_h, ro_w1, ro_b1, ro_w2, ro_b2, ro_w3, ro_b3, out);
}
